// ITERSTEP_84920093377197
// MI455X (gfx1250) — matrix operand order repaired and checked by reading, not compiled, not run on hardware
//
#include <hip/hip_runtime.h>
#include <hip/hip_bf16.h>
#include <stdint.h>

// Problem constants (from reference): B=8, N=4096, T=16, E=65536, D=768, H=4, O=256.
#define NB 8
#define NN 4096
#define NT 16
#define NE 65536
#define ND 768
#define NH 4
#define NHD 3072      // H*D
#define PT 256        // nodes per turn
#define NO 256
#define NKC 24        // K chunks of 32 over D=768
#define NTILE 192     // 16-col tiles over HD=3072

typedef __attribute__((ext_vector_type(16))) __bf16 v16bf;
typedef __attribute__((ext_vector_type(8)))  float  v8f;
typedef __attribute__((ext_vector_type(2)))  unsigned u32x2;
typedef __attribute__((ext_vector_type(4)))  unsigned u32x4;
typedef __attribute__((ext_vector_type(8)))  unsigned u32x8;

struct Frag32 { u32x4 lo; u32x4 hi; };   // 32B, bit_cast-able to v16bf

static __device__ __forceinline__ unsigned short f2bfu(float f) {
  unsigned u = __builtin_bit_cast(unsigned, f);
  unsigned r = u + 0x7fffu + ((u >> 16) & 1u);   // round-to-nearest-even
  return (unsigned short)(r >> 16);
}
static __device__ __forceinline__ float bf2f(unsigned short s) {
  return __builtin_bit_cast(float, (unsigned)s << 16);
}
static __device__ __forceinline__ float leaky(float x) { return x > 0.f ? x : 0.2f * x; }

static __device__ __forceinline__ float waveRedSum(float v) {
#pragma unroll
  for (int m = 16; m > 0; m >>= 1) v += __shfl_xor(v, m, 32);
  return v;
}

// ---------------- one-time setup kernels ----------------

// Pre-swizzle W (f32 [768][3072]) into the B-fragment order used here (contiguous k, same as A):
// Wfrag[ntile][kc][lane][j] = bf16( W[kc*32 + half*16 + j][ntile*16 + r] )
__global__ void k_prep_w(const float* __restrict__ W, unsigned short* __restrict__ Wfrag) {
  long idx = (long)blockIdx.x * 256 + threadIdx.x;
  if (idx >= (long)NTILE * NKC * 32 * 16) return;
  int j = (int)(idx & 15);
  int lane = (int)((idx >> 4) & 31);
  long g = idx >> 9;
  int kc = (int)(g % NKC);
  int ntile = (int)(g / NKC);
  int half = lane >> 4, r = lane & 15;
  int k = kc * 32 + half * 16 + j;
  int n = ntile * 16 + r;
  Wfrag[idx] = f2bfu(W[(long)k * NHD + n]);
}

__global__ void k_init_x(const float* __restrict__ x, float* __restrict__ xcF,
                         unsigned short* __restrict__ xc16) {
  long idx = (long)blockIdx.x * 256 + threadIdx.x;
  if (idx < (long)NB * NN * ND) {
    float v = x[idx];
    xcF[idx] = v;
    xc16[idx] = f2bfu(v);
  }
}

__global__ void k_zero_cnt(int* __restrict__ cnt) {
  int idx = blockIdx.x * 256 + threadIdx.x;
  if (idx < NB * NN) cnt[idx] = 0;
}

__global__ void k_count_edges(const int* __restrict__ ei, int* __restrict__ cnt) {
  long idx = (long)blockIdx.x * 256 + threadIdx.x;
  if (idx < (long)NB * NE) {
    int b = (int)(idx / NE), e = (int)(idx % NE);
    int dst = ei[(long)b * 2 * NE + NE + e];
    atomicAdd(&cnt[b * NN + dst], 1);
  }
}

__global__ void k_scan(const int* __restrict__ cnt, int* __restrict__ roff,
                       int* __restrict__ cursor) {
  int b = blockIdx.x, tid = threadIdx.x;
  __shared__ int part[256];
  __shared__ int pre[257];
  int s = 0;
#pragma unroll
  for (int j = 0; j < 16; ++j) s += cnt[b * NN + tid * 16 + j];
  part[tid] = s;
  __syncthreads();
  if (tid == 0) {
    int run = 0;
    for (int k = 0; k < 256; ++k) { pre[k] = run; run += part[k]; }
    pre[256] = run;
  }
  __syncthreads();
  int run = pre[tid];
#pragma unroll
  for (int j = 0; j < 16; ++j) {
    int node = tid * 16 + j;
    roff[b * (NN + 1) + node] = run;
    cursor[b * NN + node] = run;
    run += cnt[b * NN + node];
  }
  if (tid == 255) roff[b * (NN + 1) + NN] = pre[256];
}

__global__ void k_scatter(const int* __restrict__ ei, int* __restrict__ cursor,
                          int* __restrict__ csrSrc) {
  long idx = (long)blockIdx.x * 256 + threadIdx.x;
  if (idx < (long)NB * NE) {
    int b = (int)(idx / NE), e = (int)(idx % NE);
    int src = ei[(long)b * 2 * NE + e];
    int dst = ei[(long)b * 2 * NE + NE + e];
    int pos = atomicAdd(&cursor[b * NN + dst], 1);
    csrSrc[(long)b * NE + pos] = src;
  }
}

// ---------------- per-turn kernels ----------------

__global__ void k_zero_att(float* __restrict__ a, float* __restrict__ d) {
  int idx = blockIdx.x * 256 + threadIdx.x;
  if (idx < NB * NN * NH) { a[idx] = 0.f; d[idx] = 0.f; }
}

// h16[0:nA] = xc16[0:nA] @ W (bf16 WMMA, f32 accum) + fused asrc/adst partials.
// Block = 256 threads = 8 waves; tile M=32 x N=256; waves 2(M) x 4(N), each wave
// owns FOUR 16x16 accumulators. A tile double-buffered in LDS (1 barrier/K-step);
// B fragments from pre-swizzled L2-resident Wfrag, software-pipelined one K-step
// ahead so WMMAs never wait on an in-flight L2 access.
__global__ void __launch_bounds__(256)
k_gemm(const unsigned short* __restrict__ A, const unsigned short* __restrict__ Wfrag,
       unsigned short* __restrict__ h16,
       const float* __restrict__ a_src, const float* __restrict__ a_dst,
       float* __restrict__ asrcB, float* __restrict__ adstB) {
  __shared__ __align__(16) unsigned short As[2][32 * 32];   // double-buffered A tile
  const int b  = blockIdx.z;
  const int m0 = blockIdx.y * 32;
  const int n0 = blockIdx.x * 256;
  const unsigned short* Ag = A + (long)b * NN * ND;
  const int tid  = threadIdx.x;
  const int wave = tid >> 5, lane = tid & 31;
  const int half = lane >> 4, r = lane & 15;
  const int wm = wave >> 2, wn = wave & 3;
  const int nt0 = (n0 >> 4) + wn * 4;              // this wave's four 16-col tiles
  const unsigned short* wf0 = Wfrag + (((long)(nt0 + 0) * NKC) * 32 + lane) * 16;
  const unsigned short* wf1 = Wfrag + (((long)(nt0 + 1) * NKC) * 32 + lane) * 16;
  const unsigned short* wf2 = Wfrag + (((long)(nt0 + 2) * NKC) * 32 + lane) * 16;
  const unsigned short* wf3 = Wfrag + (((long)(nt0 + 3) * NKC) * 32 + lane) * 16;

  const int ar = tid >> 3, ac = (tid & 7) * 4;     // A staging: 4 bf16 per thread
  const long aOff = (long)(m0 + ar) * ND + ac;
  const int arow = (wm * 16 + r) * 32;

  v8f acc0 = {}, acc1 = {}, acc2 = {}, acc3 = {};
  // prologue: B fragments for kc=0 and A tiles kc=0 (to LDS) / kc=1 (in flight)
  u32x8 b0 = *(const u32x8*)wf0;
  u32x8 b1 = *(const u32x8*)wf1;
  u32x8 b2 = *(const u32x8*)wf2;
  u32x8 b3 = *(const u32x8*)wf3;
  u32x2 areg = *(const u32x2*)(Ag + aOff);
  *(u32x2*)&As[0][ar * 32 + ac] = areg;
  areg = *(const u32x2*)(Ag + aOff + 32);
  __syncthreads();
#pragma unroll
  for (int kc = 0; kc < NKC; ++kc) {
    const int cur = kc & 1, nxt = cur ^ 1;
    u32x8 n0f, n1f, n2f, n3f;
    if (kc + 1 < NKC) {                            // issue next-step loads early
      *(u32x2*)&As[nxt][ar * 32 + ac] = areg;
      const long wko = (long)(kc + 1) * 32 * 16;
      n0f = *(const u32x8*)(wf0 + wko);
      n1f = *(const u32x8*)(wf1 + wko);
      n2f = *(const u32x8*)(wf2 + wko);
      n3f = *(const u32x8*)(wf3 + wko);
    }
    if (kc + 2 < NKC) areg = *(const u32x2*)(Ag + aOff + (kc + 2) * 32);
    Frag32 at;
    at.lo = *(const u32x4*)&As[cur][arow + half * 16];       // K runs [half*16, +8)
    at.hi = *(const u32x4*)&As[cur][arow + half * 16 + 8];   // and [half*16+8, +8)
    v16bf af = __builtin_bit_cast(v16bf, at);
    acc0 = __builtin_amdgcn_wmma_f32_16x16x32_bf16(false, af, false,
             __builtin_bit_cast(v16bf, b0), (short)0, acc0, false, false);
    acc1 = __builtin_amdgcn_wmma_f32_16x16x32_bf16(false, af, false,
             __builtin_bit_cast(v16bf, b1), (short)0, acc1, false, false);
    acc2 = __builtin_amdgcn_wmma_f32_16x16x32_bf16(false, af, false,
             __builtin_bit_cast(v16bf, b2), (short)0, acc2, false, false);
    acc3 = __builtin_amdgcn_wmma_f32_16x16x32_bf16(false, af, false,
             __builtin_bit_cast(v16bf, b3), (short)0, acc3, false, false);
    if (kc + 1 < NKC) { b0 = n0f; b1 = n1f; b2 = n2f; b3 = n3f; }  // renamed on unroll
    __syncthreads();
  }
  // epilogue: bf16 h stores + fused attention-dot partials (one head per block)
  const int colB = n0 + wn * 64 + r;
  const int hh = n0 / ND;
  const int dB = colB - hh * ND;
  float ws[4], wd[4];
#pragma unroll
  for (int s = 0; s < 4; ++s) {
    ws[s] = a_src[hh * ND + dB + s * 16];
    wd[s] = a_dst[hh * ND + dB + s * 16];
  }
#pragma unroll
  for (int v = 0; v < 8; ++v) {
    const int row = m0 + wm * 16 + v + half * 8;
    const long rb = ((long)b * NN + row) * NHD;
    h16[rb + colB +  0] = f2bfu(acc0[v]);
    h16[rb + colB + 16] = f2bfu(acc1[v]);
    h16[rb + colB + 32] = f2bfu(acc2[v]);
    h16[rb + colB + 48] = f2bfu(acc3[v]);
    float pa = acc0[v] * ws[0] + acc1[v] * ws[1] + acc2[v] * ws[2] + acc3[v] * ws[3];
    float pd = acc0[v] * wd[0] + acc1[v] * wd[1] + acc2[v] * wd[2] + acc3[v] * wd[3];
#pragma unroll
    for (int m = 1; m < 16; m <<= 1) {             // reduce over 16 cols (half-wave)
      pa += __shfl_xor(pa, m, 32);
      pd += __shfl_xor(pd, m, 32);
    }
    if (r == 0) {
      atomicAdd(&asrcB[((long)b * NN + row) * NH + hh], pa);
      atomicAdd(&adstB[((long)b * NN + row) * NH + hh], pd);
    }
  }
}

// active nodes with turn < t: xn = mean_heads(h) + b_gat
__global__ void k_oldupdate(const unsigned short* __restrict__ h16,
                            const float* __restrict__ b_gat,
                            float* __restrict__ xcF, unsigned short* __restrict__ xc16,
                            int nOld) {
  long idx = (long)blockIdx.x * 256 + threadIdx.x;
  if (idx >= (long)NB * nOld * ND) return;
  int d = (int)(idx % ND);
  int i = (int)((idx / ND) % nOld);
  int b = (int)(idx / ((long)nOld * ND));
  const unsigned short* hr = h16 + ((long)b * NN + i) * NHD;
  float v = 0.25f * (bf2f(hr[d]) + bf2f(hr[ND + d]) +
                     bf2f(hr[2 * ND + d]) + bf2f(hr[3 * ND + d])) + b_gat[d];
  long o = ((long)b * NN + i) * ND + d;
  xcF[o] = v;
  xc16[o] = f2bfu(v);
}

// dst nodes of current turn: softmax aggregation over incoming edges + self loop.
__global__ void __launch_bounds__(256)
k_dstattn(const unsigned short* __restrict__ h16, const float* __restrict__ asrcB,
          const float* __restrict__ adstB, const int* __restrict__ roff,
          const int* __restrict__ csrSrc, const float* __restrict__ b_gat,
          float* __restrict__ xcF, unsigned short* __restrict__ xc16, int t) {
  const int b = blockIdx.y;
  const int i = t * PT + blockIdx.x;
  const int tid = threadIdx.x;
  __shared__ float msg[NH][ND];
  __shared__ float red[NH][256];
  __shared__ float mM[NH];
  __shared__ float dden[NH];
  const float* as = asrcB + (long)b * NN * NH;
  const float* ad = adstB + (long)b * NN * NH;
  float adi[NH], sh[NH];
#pragma unroll
  for (int hh = 0; hh < NH; ++hh) {
    adi[hh] = ad[i * NH + hh];
    sh[hh] = leaky(as[i * NH + hh] + adi[hh]);
  }
  const int e0 = roff[b * (NN + 1) + i];
  const int e1 = roff[b * (NN + 1) + i + 1];
  const int* srcs = csrSrc + (long)b * NE;
  float lm[NH];
#pragma unroll
  for (int hh = 0; hh < NH; ++hh) lm[hh] = -3.0e38f;
  for (int e = e0 + tid; e < e1; e += 256) {
    int s = srcs[e];
#pragma unroll
    for (int hh = 0; hh < NH; ++hh)
      lm[hh] = fmaxf(lm[hh], leaky(as[s * NH + hh] + adi[hh]));
  }
#pragma unroll
  for (int hh = 0; hh < NH; ++hh) red[hh][tid] = lm[hh];
  __syncthreads();
  for (int sft = 128; sft > 0; sft >>= 1) {
    if (tid < sft)
#pragma unroll
      for (int hh = 0; hh < NH; ++hh)
        red[hh][tid] = fmaxf(red[hh][tid], red[hh][tid + sft]);
    __syncthreads();
  }
  if (tid == 0)
#pragma unroll
    for (int hh = 0; hh < NH; ++hh) mM[hh] = fmaxf(red[hh][0], sh[hh]);
  __syncthreads();
  float mr[NH];
#pragma unroll
  for (int hh = 0; hh < NH; ++hh) mr[hh] = mM[hh];
  for (int idx = tid; idx < NH * ND; idx += 256) (&msg[0][0])[idx] = 0.f;
  __syncthreads();
  const unsigned short* hbase = h16 + (long)b * NN * NHD;
  float dloc[NH] = {0.f, 0.f, 0.f, 0.f};
  for (int e = e0; e < e1; ++e) {                  // serial edges, parallel features
    int s = srcs[e];
    float w[NH];
#pragma unroll
    for (int hh = 0; hh < NH; ++hh)
      w[hh] = __expf(leaky(as[s * NH + hh] + adi[hh]) - mr[hh]);
    if (tid == 0)
#pragma unroll
      for (int hh = 0; hh < NH; ++hh) dloc[hh] += w[hh];
    const unsigned short* hr = hbase + (long)s * NHD;
    for (int d = tid; d < ND; d += 256)
#pragma unroll
      for (int hh = 0; hh < NH; ++hh) msg[hh][d] += w[hh] * bf2f(hr[hh * ND + d]);
  }
  float wS[NH];
#pragma unroll
  for (int hh = 0; hh < NH; ++hh) wS[hh] = __expf(sh[hh] - mr[hh]);
  const unsigned short* hi = hbase + (long)i * NHD;
  for (int d = tid; d < ND; d += 256)
#pragma unroll
    for (int hh = 0; hh < NH; ++hh) msg[hh][d] += wS[hh] * bf2f(hi[hh * ND + d]);
  if (tid == 0)
#pragma unroll
    for (int hh = 0; hh < NH; ++hh) dden[hh] = dloc[hh] + wS[hh];
  __syncthreads();
  float inv[NH];
#pragma unroll
  for (int hh = 0; hh < NH; ++hh) inv[hh] = 1.f / dden[hh];
  for (int d = tid; d < ND; d += 256) {
    float o = 0.25f * (msg[0][d] * inv[0] + msg[1][d] * inv[1] +
                       msg[2][d] * inv[2] + msg[3][d] * inv[3]) + b_gat[d];
    long off = ((long)b * NN + i) * ND + d;
    xcF[off] = o;
    xc16[off] = f2bfu(o);
  }
}

__global__ void __launch_bounds__(256)
k_gate(const float* __restrict__ xcF, const float* __restrict__ w_att,
       const float* __restrict__ b_att, float* __restrict__ gate, int nA) {
  int wave = (blockIdx.x * 256 + threadIdx.x) >> 5;
  int lane = threadIdx.x & 31;
  if (wave >= NB * nA) return;
  int b = wave / nA, i = wave % nA;
  const float* xr = xcF + ((long)b * NN + i) * ND;
  float s = 0.f;
  for (int d = lane; d < ND; d += 32) s += xr[d] * w_att[d];
  s = waveRedSum(s);
  if (lane == 0) gate[b * NN + i] = s + b_att[0];
}

__global__ void k_softmax(float* __restrict__ gate, int nA) {
  int b = blockIdx.x, tid = threadIdx.x;
  __shared__ float rr[256];
  float lm = -3.0e38f;
  for (int i = tid; i < nA; i += 256) lm = fmaxf(lm, gate[b * NN + i]);
  rr[tid] = lm;
  __syncthreads();
  for (int s = 128; s > 0; s >>= 1) {
    if (tid < s) rr[tid] = fmaxf(rr[tid], rr[tid + s]);
    __syncthreads();
  }
  float gm = rr[0];
  __syncthreads();
  float ls = 0.f;
  for (int i = tid; i < nA; i += 256) ls += __expf(gate[b * NN + i] - gm);
  rr[tid] = ls;
  __syncthreads();
  for (int s = 128; s > 0; s >>= 1) {
    if (tid < s) rr[tid] += rr[tid + s];
    __syncthreads();
  }
  float inv = 1.f / rr[0];
  __syncthreads();
  for (int i = tid; i < nA; i += 256)
    gate[b * NN + i] = __expf(gate[b * NN + i] - gm) * inv;
}

__global__ void k_pooled(const float* __restrict__ gate, const float* __restrict__ xcF,
                         float* __restrict__ pooled, int nA) {
  int b = blockIdx.y;
  int d = blockIdx.x * 256 + threadIdx.x;
  float acc = 0.f;
  const float* xg = xcF + (long)b * NN * ND;
  const float* gg = gate + b * NN;
  for (int i = 0; i < nA; ++i) acc += gg[i] * xg[(long)i * ND + d];
  pooled[b * ND + d] = acc;
}

__global__ void k_classify(const float* __restrict__ pooled, const float* __restrict__ w_cls,
                           const float* __restrict__ b_cls, float* __restrict__ out, int t) {
  int b = blockIdx.x, o = threadIdx.x;
  float acc = b_cls[o];
  const float* p = pooled + b * ND;
  for (int d = 0; d < ND; ++d) acc += p[d] * w_cls[d * NO + o];
  out[((long)b * NT + t) * NO + o] = acc;
}

// ---------------- host orchestration ----------------

extern "C" void kernel_launch(void* const* d_in, const int* in_sizes, int n_in,
                              void* d_out, int out_size, void* d_ws, size_t ws_size,
                              hipStream_t stream) {
  const float* x      = (const float*)d_in[0];
  const int*   ei     = (const int*)d_in[1];
  (void)d_in[2]; // turns[i] == i / 256 by construction (nodes sorted by turn)
  const float* W      = (const float*)d_in[3];
  const float* a_src  = (const float*)d_in[4];
  const float* a_dst  = (const float*)d_in[5];
  const float* b_gat  = (const float*)d_in[6];
  const float* w_att  = (const float*)d_in[7];
  const float* b_att  = (const float*)d_in[8];
  const float* w_cls  = (const float*)d_in[9];
  const float* b_cls  = (const float*)d_in[10];
  float* out = (float*)d_out;
  (void)in_sizes; (void)n_in; (void)out_size; (void)ws_size;

  uint8_t* p = (uint8_t*)d_ws;
  auto alloc = [&](size_t bytes) -> void* {
    void* r = (void*)p;
    p += (bytes + 255) & ~(size_t)255;
    return r;
  };
  float*          xcF   = (float*)alloc((size_t)NB * NN * ND * 4);          //  96 MB
  unsigned short* xc16  = (unsigned short*)alloc((size_t)NB * NN * ND * 2); //  48 MB
  unsigned short* Wfrag = (unsigned short*)alloc((size_t)ND * NHD * 2);     // 4.7 MB
  unsigned short* h16   = (unsigned short*)alloc((size_t)NB * NN * NHD * 2);// 192 MB
  float*          asrcB = (float*)alloc((size_t)NB * NN * NH * 4);
  float*          adstB = (float*)alloc((size_t)NB * NN * NH * 4);
  float*          gate  = (float*)alloc((size_t)NB * NN * 4);
  float*          pooled= (float*)alloc((size_t)NB * ND * 4);
  int*            cnt   = (int*)alloc((size_t)NB * NN * 4);
  int*            roff  = (int*)alloc((size_t)NB * (NN + 1) * 4);
  int*            cursor= (int*)alloc((size_t)NB * NN * 4);
  int*            csrSrc= (int*)alloc((size_t)NB * NE * 4);

  // one-time setup
  k_prep_w  <<<(ND * NHD + 255) / 256, 256, 0, stream>>>(W, Wfrag);
  k_init_x  <<<(int)(((long)NB * NN * ND + 255) / 256), 256, 0, stream>>>(x, xcF, xc16);
  k_zero_cnt<<<(NB * NN + 255) / 256, 256, 0, stream>>>(cnt);
  k_count_edges<<<(int)(((long)NB * NE + 255) / 256), 256, 0, stream>>>(ei, cnt);
  k_scan    <<<NB, 256, 0, stream>>>(cnt, roff, cursor);
  k_scatter <<<(int)(((long)NB * NE + 255) / 256), 256, 0, stream>>>(ei, cursor, csrSrc);

  for (int t = 0; t < NT; ++t) {
    const int nA = (t + 1) * PT;
    const int nOld = t * PT;

    k_zero_att<<<(NB * NN * NH + 255) / 256, 256, 0, stream>>>(asrcB, adstB);

    dim3 gg(NHD / 256, nA / 32, NB);
    k_gemm<<<gg, 256, 0, stream>>>(xc16, Wfrag, h16, a_src, a_dst, asrcB, adstB);

    if (nOld > 0)
      k_oldupdate<<<(int)(((long)NB * nOld * ND + 255) / 256), 256, 0, stream>>>(
          h16, b_gat, xcF, xc16, nOld);

    dim3 ga(PT, NB);
    k_dstattn<<<ga, 256, 0, stream>>>(h16, asrcB, adstB, roff, csrSrc, b_gat, xcF, xc16, t);

    int gwaves = NB * nA;
    k_gate<<<(gwaves + 7) / 8, 256, 0, stream>>>(xcF, w_att, b_att, gate, nA);
    k_softmax<<<NB, 256, 0, stream>>>(gate, nA);
    dim3 gp(ND / 256, NB);
    k_pooled<<<gp, 256, 0, stream>>>(gate, xcF, pooled, nA);
    k_classify<<<NB, NO, 0, stream>>>(pooled, w_cls, b_cls, out, t);
  }
}
